// GATLayer_48112223650337
// MI455X (gfx1250) — matrix operand order repaired and checked by reading, compiled, not run on hardware
//
#include <hip/hip_runtime.h>

#define B_ 8
#define N_ 2048
#define F_ 256
#define ALPHA_ 0.2f
#define NEG_INF_ -9.0e15f

typedef __attribute__((ext_vector_type(16))) __bf16 v16bf;
typedef __attribute__((ext_vector_type(8)))  float  v8f;

union Frag16 { v16bf v; unsigned int u[8]; };

#if defined(__HIP_DEVICE_COMPILE__) && \
    __has_builtin(__builtin_amdgcn_global_load_async_to_lds_b128) && \
    __has_builtin(__builtin_amdgcn_s_wait_asynccnt)
#define USE_ASYNC 1
#else
#define USE_ASYNC 0
#endif

#if USE_ASYNC
// Prototype (from clang diagnostic): param 0 is `int4 addrspace(1)*`,
// param 1 is the LDS destination, then i32 offset, i32 cpol.
typedef int v4i_async __attribute__((vector_size(16)));
typedef __attribute__((address_space(1))) v4i_async* gasync_p;
typedef __attribute__((address_space(3))) v4i_async* lasync_p;
// Global generic pointers are numerically identical to as(1); LDS generic
// pointers carry the LDS byte offset in their low 32 bits (aperture rule),
// so integer round-trips give valid segment pointers.
__device__ __forceinline__ void async_cp16(const void* g, void* l) {
    __builtin_amdgcn_global_load_async_to_lds_b128(
        (gasync_p)(uintptr_t)g,
        (lasync_p)(unsigned int)(uintptr_t)l, 0, 0);
}
__device__ __forceinline__ void async_fence() { __builtin_amdgcn_s_wait_asynccnt(0); }
#else
__device__ __forceinline__ void async_cp16(const void* g, void* l) {
    *(uint4*)l = *(const uint4*)g;
}
__device__ __forceinline__ void async_fence() {}
#endif

__device__ __forceinline__ unsigned short f2bf(float f) {
    unsigned int u = __float_as_uint(f);
    unsigned int r = (u + 0x7FFFu + ((u >> 16) & 1u)) >> 16;   // RNE
    return (unsigned short)r;
}

// ---------------------------------------------------------------- kernel 0
// WbfT[out][k] = bf16(W[k][out])  (transposed so B-tiles stage as [col][k])
__global__ __launch_bounds__(256)
void k_convert_wT(const float* __restrict__ W, unsigned short* __restrict__ WbfT) {
    int i = blockIdx.x * 256 + threadIdx.x;     // 0 .. 65535
    int k = i & (F_ - 1), o = i >> 8;
    WbfT[(size_t)o * F_ + k] = f2bf(W[(size_t)k * F_ + o]);
}

// ---------------------------------------------------------------- kernel 1
// h = x @ W via bf16 WMMA. 32 n-rows x 256 fout per block; 8 waves,
// 4 accumulators each (2 row-tiles x 2 col-tiles). Writes h f32 (d_out as
// scratch) and transposed bf16 h (hbfT[fout][m]) for the attention GEMM.
__global__ __launch_bounds__(256)
void k_h_gemm(const float* __restrict__ x, const unsigned short* __restrict__ WbfT,
              float* __restrict__ h_out, unsigned short* __restrict__ hbfT) {
    __shared__ __align__(16) unsigned short ldsWT[F_][32]; // 16KB  [out][k]
    __shared__ __align__(16) unsigned short ldsA[32][32];  // 2KB   [n][k]
    const int t = threadIdx.x;
    const int b = blockIdx.y;
    const int n0 = blockIdx.x * 32;
    const int lane = t & 31, wv = t >> 5;
    const int half = lane >> 4, mrow = lane & 15;
    const float* xb = x + ((size_t)b * N_ + n0) * (size_t)F_;
    v8f acc00 = {}, acc01 = {}, acc10 = {}, acc11 = {};

    for (int k0 = 0; k0 < F_; k0 += 32) {
        // stage W^T tile [256 out][32 k] : 1024 x 16B, 4 segments/thread
        #pragma unroll
        for (int j = 0; j < 4; ++j) {
            int i = t + j * 256;
            int fo = i >> 2, seg = i & 3;
            async_cp16(WbfT + (size_t)fo * F_ + k0 + seg * 8, &ldsWT[fo][seg * 8]);
        }
        // stage A tile: x[n0..n0+31][k0..k0+31] -> bf16, 4 elems/thread
        {
            int idx = t * 4;
            int r = idx >> 5, kk = idx & 31;
            float4 xv = *(const float4*)&xb[(size_t)r * F_ + k0 + kk];
            unsigned int lo = (unsigned)f2bf(xv.x) | ((unsigned)f2bf(xv.y) << 16);
            unsigned int hi = (unsigned)f2bf(xv.z) | ((unsigned)f2bf(xv.w) << 16);
            *(uint2*)&ldsA[r][kk] = make_uint2(lo, hi);
        }
        async_fence();
        __syncthreads();
        // A fragments (ISA bf16 A 16x32 lane layout), rows mrow / mrow+16
        Frag16 a0, a1;
        #pragma unroll
        for (int vi = 0; vi < 8; ++vi) {
            int kb = ((vi & 3) * 2) + ((vi >> 2) * 16) + half * 8;
            a0.u[vi] = *(const unsigned int*)&ldsA[mrow][kb];
            a1.u[vi] = *(const unsigned int*)&ldsA[16 + mrow][kb];
        }
        // B fragments: transposed LDS -> one aligned b32 per K-pair
        Frag16 bf0, bf1;
        int col0 = wv * 32 + mrow, col1 = col0 + 16;
        #pragma unroll
        for (int vi = 0; vi < 8; ++vi) {
            int kr = ((vi & 3) * 2) + ((vi >> 2) * 16) + half * 8;
            bf0.u[vi] = *(const unsigned int*)&ldsWT[col0][kr];
            bf1.u[vi] = *(const unsigned int*)&ldsWT[col1][kr];
        }
        acc00 = __builtin_amdgcn_wmma_f32_16x16x32_bf16(false, a0.v, false, bf0.v, (short)0, acc00, false, false);
        acc01 = __builtin_amdgcn_wmma_f32_16x16x32_bf16(false, a0.v, false, bf1.v, (short)0, acc01, false, false);
        acc10 = __builtin_amdgcn_wmma_f32_16x16x32_bf16(false, a1.v, false, bf0.v, (short)0, acc10, false, false);
        acc11 = __builtin_amdgcn_wmma_f32_16x16x32_bf16(false, a1.v, false, bf1.v, (short)0, acc11, false, false);
        __syncthreads();
    }
    #pragma unroll
    for (int r = 0; r < 8; ++r) {
        int row  = r + 8 * half;
        int col0 = wv * 32 + mrow;
        size_t o0 = (((size_t)b * N_) + n0 + row) * (size_t)F_ + col0;
        size_t o1 = o0 + (size_t)16 * F_;                       // rows +16
        h_out[o0]      = acc00[r];  h_out[o0 + 16] = acc01[r];
        h_out[o1]      = acc10[r];  h_out[o1 + 16] = acc11[r];
        size_t tb = ((size_t)b * F_ + col0) * (size_t)N_ + n0 + row;
        hbfT[tb]                        = f2bf(acc00[r]);
        hbfT[tb + (size_t)16 * N_]      = f2bf(acc01[r]);
        hbfT[tb + 16]                   = f2bf(acc10[r]);
        hbfT[tb + (size_t)16 * N_ + 16] = f2bf(acc11[r]);
    }
}

// ---------------------------------------------------------------- kernel 2
// e_src/e_dst row dots, 1 wave per row.
__global__ __launch_bounds__(256)
void k_edots(const float* __restrict__ h, const float* __restrict__ a,
             float* __restrict__ e_src, float* __restrict__ e_dst) {
    int lane = threadIdx.x & 31;
    int row  = blockIdx.x * 8 + (threadIdx.x >> 5);
    const float* hr = h + (size_t)row * F_;
    float s0 = 0.f, s1 = 0.f;
    for (int j = lane; j < F_; j += 32) {
        float hv = hr[j];
        s0 += hv * a[j];
        s1 += hv * a[F_ + j];
    }
    #pragma unroll
    for (int off = 16; off > 0; off >>= 1) {
        s0 += __shfl_down(s0, off, 32);
        s1 += __shfl_down(s1, off, 32);
    }
    if (lane == 0) { e_src[row] = s0; e_dst[row] = s1; }
}

// ---------------------------------------------------------------- kernel 3
// Masked-softmax row stats (finite NEG_INF reproduces the reference's
// uniform-softmax behavior for fully masked rows).
__global__ __launch_bounds__(256)
void k_rowstats(const int* __restrict__ adj, const float* __restrict__ e_src,
                const float* __restrict__ e_dst, float* __restrict__ rowmax,
                float* __restrict__ rowsum) {
    __shared__ float s[N_];
    __shared__ float red[256];
    int t = threadIdx.x;
    int n = blockIdx.x, b = blockIdx.y;
    const int* arow = adj + ((size_t)b * N_ + n) * (size_t)N_;
    float es = e_src[b * N_ + n];
    float lmax = NEG_INF_;
    for (int m = t; m < N_; m += 256) {
        float v = es + e_dst[b * N_ + m];
        v = v > 0.f ? v : ALPHA_ * v;
        v = (arow[m] > 0) ? v : NEG_INF_;
        s[m] = v;
        lmax = fmaxf(lmax, v);
    }
    red[t] = lmax;
    __syncthreads();
    for (int off = 128; off > 0; off >>= 1) {
        if (t < off) red[t] = fmaxf(red[t], red[t + off]);
        __syncthreads();
    }
    float mx = red[0];
    __syncthreads();
    float lsum = 0.f;
    for (int m = t; m < N_; m += 256) lsum += __expf(s[m] - mx);
    red[t] = lsum;
    __syncthreads();
    for (int off = 128; off > 0; off >>= 1) {
        if (t < off) red[t] += red[t + off];
        __syncthreads();
    }
    if (t == 0) { rowmax[b * N_ + n] = mx; rowsum[b * N_ + n] = red[0]; }
}

// ---------------------------------------------------------------- kernel 4
// h' = softmax(scores) @ h via bf16 WMMA; probabilities regenerated on the
// fly (unnormalized, /rowsum + ELU in epilogue). 32 n-rows x 256 fout per
// block; 4 WMMAs per wave per step.
__global__ __launch_bounds__(256)
void k_aggregate(const int* __restrict__ adj, const unsigned short* __restrict__ hbfT,
                 const float* __restrict__ e_src, const float* __restrict__ e_dst,
                 const float* __restrict__ rowmax, const float* __restrict__ rowsum,
                 float* __restrict__ out) {
    __shared__ __align__(16) unsigned short ldsHT[F_][32];  // 16KB [fout][m-k]
    __shared__ __align__(16) unsigned short ldsP[32][32];   // 2KB  [n][m-k]
    __shared__ float ls_es[32], ls_mx[32], ls_rs[32];
    int t = threadIdx.x;
    int b = blockIdx.y;
    int n0 = blockIdx.x * 32;
    if (t < 32) {
        int row = b * N_ + n0 + t;
        ls_es[t] = e_src[row];
        ls_mx[t] = rowmax[row];
        ls_rs[t] = rowsum[row];
    }
    int lane = t & 31, wv = t >> 5;
    int half = lane >> 4, mrow = lane & 15;
    v8f acc00 = {}, acc01 = {}, acc10 = {}, acc11 = {};
    __syncthreads();

    // P-tile thread mapping: 4 consecutive m per thread
    const int pr = (t * 4) >> 5, pm = (t * 4) & 31;
    const int* arow = adj + (((size_t)b * N_) + n0 + pr) * (size_t)N_ + pm;

    for (int m0 = 0; m0 < N_; m0 += 32) {
        // stage h^T tile [256 fout][32 m] : 1024 x 16B, async -> LDS
        #pragma unroll
        for (int j = 0; j < 4; ++j) {
            int i = t + j * 256;
            int fo = i >> 2, seg = i & 3;
            async_cp16(hbfT + ((size_t)b * F_ + fo) * (size_t)N_ + m0 + seg * 8,
                       &ldsHT[fo][seg * 8]);
        }
        // P tile: p = exp(mask(leakyrelu(es+ed)) - rowmax), bf16
        {
            int4   a4 = *(const int4*)&arow[m0];
            float4 ed = *(const float4*)&e_dst[b * N_ + m0 + pm];
            if (m0 + 32 < N_) __builtin_prefetch(&arow[m0 + 32], 0, 0);
            float es = ls_es[pr], mx = ls_mx[pr];
            float p[4];
            float e0 = es + ed.x, e1 = es + ed.y, e2 = es + ed.z, e3 = es + ed.w;
            e0 = e0 > 0.f ? e0 : ALPHA_ * e0;  e1 = e1 > 0.f ? e1 : ALPHA_ * e1;
            e2 = e2 > 0.f ? e2 : ALPHA_ * e2;  e3 = e3 > 0.f ? e3 : ALPHA_ * e3;
            p[0] = __expf((a4.x > 0 ? e0 : NEG_INF_) - mx);
            p[1] = __expf((a4.y > 0 ? e1 : NEG_INF_) - mx);
            p[2] = __expf((a4.z > 0 ? e2 : NEG_INF_) - mx);
            p[3] = __expf((a4.w > 0 ? e3 : NEG_INF_) - mx);
            unsigned int lo = (unsigned)f2bf(p[0]) | ((unsigned)f2bf(p[1]) << 16);
            unsigned int hi = (unsigned)f2bf(p[2]) | ((unsigned)f2bf(p[3]) << 16);
            *(uint2*)&ldsP[pr][pm] = make_uint2(lo, hi);
        }
        async_fence();
        __syncthreads();
        Frag16 a0, a1;
        #pragma unroll
        for (int vi = 0; vi < 8; ++vi) {
            int kb = ((vi & 3) * 2) + ((vi >> 2) * 16) + half * 8;
            a0.u[vi] = *(const unsigned int*)&ldsP[mrow][kb];
            a1.u[vi] = *(const unsigned int*)&ldsP[16 + mrow][kb];
        }
        Frag16 bf0, bf1;
        int col0 = wv * 32 + mrow, col1 = col0 + 16;
        #pragma unroll
        for (int vi = 0; vi < 8; ++vi) {
            int kr = ((vi & 3) * 2) + ((vi >> 2) * 16) + half * 8;
            bf0.u[vi] = *(const unsigned int*)&ldsHT[col0][kr];
            bf1.u[vi] = *(const unsigned int*)&ldsHT[col1][kr];
        }
        acc00 = __builtin_amdgcn_wmma_f32_16x16x32_bf16(false, a0.v, false, bf0.v, (short)0, acc00, false, false);
        acc01 = __builtin_amdgcn_wmma_f32_16x16x32_bf16(false, a0.v, false, bf1.v, (short)0, acc01, false, false);
        acc10 = __builtin_amdgcn_wmma_f32_16x16x32_bf16(false, a1.v, false, bf0.v, (short)0, acc10, false, false);
        acc11 = __builtin_amdgcn_wmma_f32_16x16x32_bf16(false, a1.v, false, bf1.v, (short)0, acc11, false, false);
        __syncthreads();
    }
    // epilogue: /rowsum, ELU, store f32
    #pragma unroll
    for (int r = 0; r < 8; ++r) {
        int row = r + 8 * half;
        float i0 = 1.0f / ls_rs[row];
        float i1 = 1.0f / ls_rs[row + 16];
        float v00 = acc00[r] * i0, v01 = acc01[r] * i0;
        float v10 = acc10[r] * i1, v11 = acc11[r] * i1;
        v00 = v00 > 0.f ? v00 : (__expf(v00) - 1.0f);
        v01 = v01 > 0.f ? v01 : (__expf(v01) - 1.0f);
        v10 = v10 > 0.f ? v10 : (__expf(v10) - 1.0f);
        v11 = v11 > 0.f ? v11 : (__expf(v11) - 1.0f);
        int col0 = wv * 32 + mrow;
        size_t o0 = (((size_t)b * N_) + n0 + row) * (size_t)F_ + col0;
        size_t o1 = o0 + (size_t)16 * F_;
        out[o0] = v00;  out[o0 + 16] = v01;
        out[o1] = v10;  out[o1 + 16] = v11;
    }
}

// ---------------------------------------------------------------- launcher
extern "C" void kernel_launch(void* const* d_in, const int* in_sizes, int n_in,
                              void* d_out, int out_size, void* d_ws, size_t ws_size,
                              hipStream_t stream) {
    (void)in_sizes; (void)n_in; (void)out_size; (void)ws_size;
    const float* x   = (const float*)d_in[0];
    const int*   adj = (const int*)d_in[1];
    const float* W   = (const float*)d_in[2];
    const float* a   = (const float*)d_in[3];
    float* out = (float*)d_out;

    char* ws = (char*)d_ws;
    unsigned short* WbfT = (unsigned short*)ws;                          // 128KB
    unsigned short* hbfT = (unsigned short*)(ws + (size_t)F_ * F_ * 2);  // 8MB
    float* e_src = (float*)(ws + (size_t)F_ * F_ * 2 + (size_t)B_ * N_ * F_ * 2);
    float* e_dst = e_src + (size_t)B_ * N_;
    float* rmax  = e_dst + (size_t)B_ * N_;
    float* rsum  = rmax  + (size_t)B_ * N_;

    k_convert_wT<<<dim3((F_ * F_) / 256), dim3(256), 0, stream>>>(W, WbfT);
    // d_out temporarily holds h (f32); overwritten by k_aggregate afterwards.
    k_h_gemm   <<<dim3(N_ / 32, B_), dim3(256), 0, stream>>>(x, WbfT, out, hbfT);
    k_edots    <<<dim3((B_ * N_) / 8), dim3(256), 0, stream>>>(out, a, e_src, e_dst);
    k_rowstats <<<dim3(N_, B_), dim3(256), 0, stream>>>(adj, e_src, e_dst, rmax, rsum);
    k_aggregate<<<dim3(N_ / 32, B_), dim3(256), 0, stream>>>(adj, hbfT, e_src, e_dst, rmax, rsum, out);
}
